// InterpretableMultiHeadAttention_89386859364957
// MI455X (gfx1250) — matrix operand order repaired and checked by reading, not compiled, not run on hardware
//
#include <hip/hip_runtime.h>
#include <hip/hip_bf16.h>

// ---------------------------------------------------------------------------
// InterpretableMultiHeadAttention for MI455X (gfx1250), wave32 + WMMA f16.
// B=4, S=2048, D=512, H=8, E=64.
// Round 2: LDS-resident attn_weights accumulation (single global writeout),
// f16 LDS prob row, pre-packed V in WMMA-B fragment order.
// ---------------------------------------------------------------------------

typedef __attribute__((ext_vector_type(16))) _Float16 v16h;
typedef __attribute__((ext_vector_type(8)))  float    v8f;

#define B_ 4
#define S_ 2048
#define D_ 512
#define H_ 8
#define E_ 64
#define NEG16 (-60000.0f)   // f16-representable stand-in for -1e9 (exp -> 0)

union AF {
    v16h         v;
    _Float16     h[16];
    unsigned int u[8];
};

__device__ __forceinline__ v8f wmma_f16(const AF& a, const AF& b, v8f c) {
    // D = A(16x32 f16) * B(32x16 f16) + C(16x16 f32)
    return __builtin_amdgcn_wmma_f32_16x16x32_f16(
        /*neg_a=*/false, a.v, /*neg_b=*/false, b.v,
        /*c_mod=*/(short)0, c, /*reuse_a=*/false, /*reuse_b=*/false);
}

// ---------------------------------------------------------------------------
// Kernel 1: f32 -> f16 convert (grid-stride)
// ---------------------------------------------------------------------------
__global__ void cvt_f32_f16(const float* __restrict__ x, _Float16* __restrict__ y, int n) {
    for (int i = blockIdx.x * blockDim.x + threadIdx.x; i < n; i += gridDim.x * blockDim.x)
        y[i] = (_Float16)x[i];
}

// ---------------------------------------------------------------------------
// Kernel 2: projection GEMM  Y[(b*nH+h)*S+s][e] = sum_d X[b*S+s][d]*W[h][e][d] + bias[h][e]
// One wave per 16x16 tile; K-loop over D_ in steps of 32 (f16 WMMA).
// A layout (16x32 f16): lane l row M=l&15; K = ((l<16)?0:16) + g*8 + 2*rr + (j&1).
// B layout (32x16 f16): lane l col N=l&15; K = ((l<16)?0:16) + j  (contiguous along D).
// ---------------------------------------------------------------------------
__global__ __launch_bounds__(32)
void proj_wmma(const _Float16* __restrict__ X,   // [B_*S_, D_]
               const _Float16* __restrict__ W,   // [nH, E_, D_]
               const float*    __restrict__ bias,// [nH, E_]
               _Float16*       __restrict__ Y,   // ((b*nH+h)*S_+s)*E_+e
               int nH) {
    const int lane = threadIdx.x & 31;
    const int m0   = blockIdx.x * 16;
    const int n0   = blockIdx.y * 16;
    const int h    = blockIdx.z;
    const int nlo  = lane & 15;
    const int akb  = (lane < 16) ? 0 : 8;
    const int bko  = (lane < 16) ? 0 : 16;

    const _Float16* xr = X + (size_t)(m0 + nlo) * D_;
    const _Float16* wr = W + ((size_t)h * E_ + n0 + nlo) * D_;

    v8f c = {};
    for (int kk = 0; kk < D_; kk += 32) {
        AF a, b;
#pragma unroll
        for (int g = 0; g < 2; ++g)
#pragma unroll
            for (int rr = 0; rr < 4; ++rr)
                a.u[g * 4 + rr] = *(const unsigned int*)(xr + kk + g * 8 + bko + 2 * rr);
#pragma unroll
        for (int j = 0; j < 8; ++j)
            b.u[j] = *(const unsigned int*)(wr + kk + bko + 2 * j);
        __builtin_prefetch(xr + kk + 64, 0, 1);
        c = wmma_f16(a, b, c);
    }

    const float bv = bias[h * E_ + n0 + nlo];
#pragma unroll
    for (int r = 0; r < 8; ++r) {
        const int m  = m0 + r + akb;
        const int bi = m / S_;
        const int s  = m % S_;
        Y[(((size_t)bi * nH + h) * S_ + s) * E_ + n0 + nlo] = (_Float16)(c[r] + bv);
    }
}

// ---------------------------------------------------------------------------
// Kernel 2b: repack projected V [B_,S_,E_] into WMMA-B fragment order:
// tile (b, t0/32, e0/16) -> 512 halves; lane l holds N=e0+(l&15),
// K=((l<16)?0:16)+j stored contiguously -> attention reads 8 dwords/lane.
// ---------------------------------------------------------------------------
__global__ __launch_bounds__(32)
void repack_vb(const _Float16* __restrict__ Vp, _Float16* __restrict__ Vb) {
    const int lane  = threadIdx.x & 31;
    const int tile  = blockIdx.x;              // b*(S_/32) + t0/32
    const int cg    = blockIdx.y;              // e0/16
    const int b     = tile / (S_ / 32);
    const int t0    = (tile % (S_ / 32)) * 32;
    const int n     = cg * 16 + (lane & 15);
    const int kbase = (lane < 16) ? 0 : 16;
    _Float16* dst = Vb + (((size_t)tile) * (E_ / 16) + cg) * 512 + lane * 16;
#pragma unroll
    for (int j = 0; j < 16; ++j)
        dst[j] = Vp[((size_t)b * S_ + t0 + kbase + j) * E_ + n];
}

// ---------------------------------------------------------------------------
// Kernel 3: fused attention. One block = (batch b, 16-query-row tile), 4 waves.
// LDS: awacc f32[16][S_] (128 KB, attn-weight accumulator, single writeout),
//      spf   f16[16][S_] ( 64 KB, score/prob row),
//      ctxs  f16[16][E_], red/rowmax/rowsum.  Total ~195 KB (CDNA5 320 KB WGP).
// ---------------------------------------------------------------------------
__global__ __launch_bounds__(128)
void attn_fused(const _Float16* __restrict__ Qp,   // [B_,H_,S_,E_]
                const _Float16* __restrict__ Kp,   // [B_,H_,S_,E_]
                const _Float16* __restrict__ Vb,   // packed B-fragment tiles
                const int*      __restrict__ mask, // [B_,S_,S_]
                const _Float16* __restrict__ Wo,   // [D_,E_]
                const float*    __restrict__ bo,   // [D_]
                float*          __restrict__ out,  // [B_,S_,D_]
                float*          __restrict__ attnw)// [B_,S_,S_]
{
    extern __shared__ char smem_raw[];
    float*    awacc  = (float*)smem_raw;               // [16][S_] 131072 B
    _Float16* spf    = (_Float16*)(awacc + 16 * S_);   // [16][S_]  65536 B
    _Float16* ctxs   = spf + 16 * S_;                  // [16][E_]   2048 B
    float*    red    = (float*)(ctxs + 16 * E_);       // [16][8]
    float*    rowmax = red + 16 * 8;                   // [16]
    float*    rowsum = rowmax + 16;                    // [16]

    const int tid  = threadIdx.x;
    const int w    = tid >> 5;
    const int lane = tid & 31;
    const int b    = blockIdx.y;
    const int row0 = blockIdx.x * 16;
    const int nlo  = lane & 15;
    const int akb  = (lane < 16) ? 0 : 8;
    const int bko  = (lane < 16) ? 0 : 16;

    v8f cacc = {};   // ctx accumulator: wave w owns E-slice [w*16, w*16+16)

    for (int h = 0; h < H_; ++h) {
        // ---- Q fragments for this head (A-matrix, two K=32 steps over E=64)
        AF aq0, aq1;
        {
            const _Float16* qr = Qp + (((size_t)(b * H_ + h)) * S_ + row0 + nlo) * E_;
#pragma unroll
            for (int g = 0; g < 2; ++g)
#pragma unroll
                for (int rr = 0; rr < 4; ++rr) {
                    aq0.u[g * 4 + rr] = *(const unsigned int*)(qr + 0  + g * 8 + bko + 2 * rr);
                    aq1.u[g * 4 + rr] = *(const unsigned int*)(qr + 32 + g * 8 + bko + 2 * rr);
                }
        }

        // ---- scores: 128 key tiles split across the 4 waves
        for (int t0 = w * 16; t0 < S_; t0 += 64) {
            const _Float16* kr = Kp + (((size_t)(b * H_ + h)) * S_ + t0 + nlo) * E_ + bko;
            __builtin_prefetch(kr + 64 * E_, 0, 1);
            AF b0, b1;
#pragma unroll
            for (int j = 0; j < 8; ++j) {
                b0.u[j] = *(const unsigned int*)(kr + 2 * j);
                b1.u[j] = *(const unsigned int*)(kr + 32 + 2 * j);
            }
            v8f c = {};
            c = wmma_f16(aq0, b0, c);
            c = wmma_f16(aq1, b1, c);
            const int t = t0 + nlo;
#pragma unroll
            for (int r = 0; r < 8; ++r) {
                const int mrow = r + akb;
                const int mk = mask[((size_t)b * S_ + row0 + mrow) * S_ + t];
                spf[mrow * S_ + t] = (_Float16)(mk ? (c[r] * 0.125f) : NEG16);
            }
        }
        __syncthreads();

        // ---- softmax over each of the 16 rows (8 threads per row)
        {
            const int r   = tid & 15;
            const int grp = tid >> 4;
            float mx = -3.4e38f;
            for (int t = grp; t < S_; t += 8) mx = fmaxf(mx, (float)spf[r * S_ + t]);
            red[r * 8 + grp] = mx;
            __syncthreads();
            if (grp == 0) {
                float m2 = red[r * 8];
#pragma unroll
                for (int g = 1; g < 8; ++g) m2 = fmaxf(m2, red[r * 8 + g]);
                rowmax[r] = m2;
            }
            __syncthreads();
            const float rm = rowmax[r];
            float sm = 0.f;
            for (int t = grp; t < S_; t += 8) {
                const float e = __expf((float)spf[r * S_ + t] - rm);
                spf[r * S_ + t] = (_Float16)e;
                sm += e;
            }
            red[r * 8 + grp] = sm;
            __syncthreads();
            if (grp == 0) {
                float s2 = 0.f;
#pragma unroll
                for (int g = 0; g < 8; ++g) s2 += red[r * 8 + g];
                rowsum[r] = s2;
            }
            __syncthreads();
            const float inv = 1.0f / rowsum[r];
            for (int t = grp; t < S_; t += 8) {
                const float p = (float)spf[r * S_ + t] * inv;
                spf[r * S_ + t] = (_Float16)p;
                if (h == 0) awacc[r * S_ + t] = p;        // LDS accumulate,
                else        awacc[r * S_ + t] += p;       // deterministic order
            }
        }
        __syncthreads();

        // ---- probs @ V : wave w owns E-slice w*16..+16; K-loop over S.
        // A-fragment: 8 dword LDS loads (consecutive K pairs contiguous in spf).
        // B-fragment: 8 dword loads from pre-packed Vb tile.
        {
            const _Float16* vtile = Vb + (((size_t)b * (S_ / 32)) * (E_ / 16) + w) * 512
                                      + lane * 16;
            for (int t0 = 0; t0 < S_; t0 += 32) {
                AF ap, bv_;
#pragma unroll
                for (int g = 0; g < 2; ++g)
#pragma unroll
                    for (int rr = 0; rr < 4; ++rr)
                        ap.u[g * 4 + rr] = *(const unsigned int*)
                            (spf + nlo * S_ + t0 + g * 8 + bko + 2 * rr);
#pragma unroll
                for (int j = 0; j < 8; ++j)
                    bv_.u[j] = *(const unsigned int*)(vtile + 2 * j);
                cacc = wmma_f16(ap, bv_, cacc);
                vtile += (size_t)(E_ / 16) * 512;   // next 32-row tile
            }
        }
        __syncthreads();   // spf reused next head
    }

    // ---- attn_weights: single global writeout (scaled 1/H)
    for (int idx = tid; idx < 16 * S_; idx += 128) {
        const int r = idx >> 11;          // /S_
        const int t = idx & (S_ - 1);
        attnw[((size_t)b * S_ + row0 + r) * S_ + t] = awacc[idx] * (1.0f / H_);
    }

    // ---- ctx averaged over heads -> LDS f16
#pragma unroll
    for (int r = 0; r < 8; ++r)
        ctxs[(r + akb) * E_ + w * 16 + nlo] = (_Float16)(cacc[r] * (1.0f / H_));
    __syncthreads();

    // ---- output projection: out = ctx @ Wo^T + bo, 32 D-tiles split 8/wave
    AF ac0, ac1;
#pragma unroll
    for (int g = 0; g < 2; ++g)
#pragma unroll
        for (int rr = 0; rr < 4; ++rr) {
            ac0.u[g * 4 + rr] = *(const unsigned int*)(ctxs + nlo * E_ + 0  + g * 8 + bko + 2 * rr);
            ac1.u[g * 4 + rr] = *(const unsigned int*)(ctxs + nlo * E_ + 32 + g * 8 + bko + 2 * rr);
        }
    for (int i = 0; i < D_ / 16 / 4; ++i) {
        const int d0 = (w * (D_ / 16 / 4) + i) * 16;
        const _Float16* wr = Wo + ((size_t)d0 + nlo) * E_ + bko;
        AF b0, b1;
#pragma unroll
        for (int j = 0; j < 8; ++j) {
            b0.u[j] = *(const unsigned int*)(wr + 2 * j);
            b1.u[j] = *(const unsigned int*)(wr + 32 + 2 * j);
        }
        v8f c = {};
        c = wmma_f16(ac0, b0, c);
        c = wmma_f16(ac1, b1, c);
        const float bv = bo[d0 + nlo];
#pragma unroll
        for (int r = 0; r < 8; ++r)
            out[((size_t)b * S_ + row0 + r + akb) * D_ + d0 + nlo] = c[r] + bv;
    }
}

// ---------------------------------------------------------------------------
// Host launcher
// ---------------------------------------------------------------------------
extern "C" void kernel_launch(void* const* d_in, const int* in_sizes, int n_in,
                              void* d_out, int out_size, void* d_ws, size_t ws_size,
                              hipStream_t stream) {
    const float* queries = (const float*)d_in[0];
    const float* keys    = (const float*)d_in[1];
    const float* values  = (const float*)d_in[2];
    const int*   mask    = (const int*)d_in[3];
    const float* Wq      = (const float*)d_in[4];
    const float* bq      = (const float*)d_in[5];
    const float* Wk      = (const float*)d_in[6];
    const float* bk      = (const float*)d_in[7];
    const float* Wv      = (const float*)d_in[8];
    const float* bv      = (const float*)d_in[9];
    const float* Wo      = (const float*)d_in[10];
    const float* bo      = (const float*)d_in[11];

    char*  ws  = (char*)d_ws;
    size_t off = 0;
    auto carve = [&](size_t bytes) {
        char* p = ws + off;
        off = (off + bytes + 255) & ~(size_t)255;
        return p;
    };
    const size_t nBSD  = (size_t)B_ * S_ * D_;
    const size_t nBHSE = (size_t)B_ * H_ * S_ * E_;
    const size_t nBSE  = (size_t)B_ * S_ * E_;
    _Float16* qh   = (_Float16*)carve(nBSD * 2);
    _Float16* kh   = (_Float16*)carve(nBSD * 2);
    _Float16* vh   = (_Float16*)carve(nBSD * 2);
    _Float16* Wq_h = (_Float16*)carve((size_t)H_ * E_ * D_ * 2);
    _Float16* Wk_h = (_Float16*)carve((size_t)H_ * E_ * D_ * 2);
    _Float16* Wv_h = (_Float16*)carve((size_t)E_ * D_ * 2);
    _Float16* Wo_h = (_Float16*)carve((size_t)D_ * E_ * 2);
    _Float16* Qp   = (_Float16*)carve(nBHSE * 2);
    _Float16* Kp   = (_Float16*)carve(nBHSE * 2);
    _Float16* Vp   = (_Float16*)carve(nBSE * 2);
    _Float16* Vb   = (_Float16*)carve(nBSE * 2);

    // f32 -> f16 converts
    cvt_f32_f16<<<dim3(1024), 256, 0, stream>>>(queries, qh, (int)nBSD);
    cvt_f32_f16<<<dim3(1024), 256, 0, stream>>>(keys,    kh, (int)nBSD);
    cvt_f32_f16<<<dim3(1024), 256, 0, stream>>>(values,  vh, (int)nBSD);
    cvt_f32_f16<<<dim3(256),  256, 0, stream>>>(Wq, Wq_h, H_ * E_ * D_);
    cvt_f32_f16<<<dim3(256),  256, 0, stream>>>(Wk, Wk_h, H_ * E_ * D_);
    cvt_f32_f16<<<dim3(64),   256, 0, stream>>>(Wv, Wv_h, E_ * D_);
    cvt_f32_f16<<<dim3(64),   256, 0, stream>>>(Wo, Wo_h, D_ * E_);

    // WMMA projections
    dim3 gq(B_ * S_ / 16, E_ / 16, H_);
    proj_wmma<<<gq, 32, 0, stream>>>(qh, Wq_h, bq, Qp, H_);
    proj_wmma<<<gq, 32, 0, stream>>>(kh, Wk_h, bk, Kp, H_);
    dim3 gv(B_ * S_ / 16, E_ / 16, 1);
    proj_wmma<<<gv, 32, 0, stream>>>(vh, Wv_h, bv, Vp, 1);

    // Repack V into WMMA-B fragment tiles
    repack_vb<<<dim3(B_ * S_ / 32, E_ / 16), 32, 0, stream>>>(Vp, Vb);

    // Fused attention (~195 KB dynamic LDS; CDNA5 WGP has 320 KB)
    const size_t smem = (size_t)16 * S_ * 4      // awacc
                      + (size_t)16 * S_ * 2      // spf
                      + (size_t)16 * E_ * 2      // ctxs
                      + 16 * 8 * 4 + 16 * 4 + 16 * 4;
    (void)hipFuncSetAttribute((const void*)attn_fused,
                              hipFuncAttributeMaxDynamicSharedMemorySize, (int)smem);
    float* out   = (float*)d_out;
    float* attnw = out + nBSD;
    attn_fused<<<dim3(S_ / 16, B_), 128, smem, stream>>>(
        Qp, Kp, Vb, mask, Wo_h, bo, out, attnw);
}
